// MultiHeadAttention_74732430950817
// MI455X (gfx1250) — matrix operand order repaired and checked by reading, not compiled, not run on hardware
//
#include <hip/hip_runtime.h>
#include <hip/hip_bf16.h>

// MHA forward for MI455X (gfx1250, wave32, WMMA).
// B=4, S=2048, E=1024, H=16, HD=64.
// Pipeline: f32->bf16 convert; QKV GEMM (WMMA bf16) with scatter epilogue into
// Q[B,H,S,HD], K[B,H,S,HD], V^T[B,H,HD,S]; flash attention (WMMA bf16, online
// softmax with WMMA row-sums + tile-max); output projection GEMM (f32 out).

constexpr int Bc  = 4;
constexpr int Sc  = 2048;
constexpr int Ec  = 1024;
constexpr int Hc  = 16;
constexpr int HDc = 64;

typedef __attribute__((ext_vector_type(16))) __bf16   v16bf;
typedef __attribute__((ext_vector_type(8)))  float    v8f;
typedef __attribute__((ext_vector_type(4)))  unsigned int uint4v;

union FragB {
    uint4v u4[2];   // two 16B chunks = 16 bf16 elements
    v16bf  bf;
};

__device__ __forceinline__ unsigned short f32_to_bf16(float f) {
    unsigned int u = __float_as_uint(f);
    unsigned int r = u + 0x7FFFu + ((u >> 16) & 1u);   // round-nearest-even
    return (unsigned short)(r >> 16);
}

__device__ __forceinline__ v8f zero8() {
    v8f z;
#pragma unroll
    for (int i = 0; i < 8; ++i) z[i] = 0.0f;
    return z;
}

// ---------------------------------------------------------------------------
// f32 -> bf16 conversion (grid-stride)
// ---------------------------------------------------------------------------
__global__ void cvt_bf16_k(const float* __restrict__ x,
                           unsigned short* __restrict__ y, int n) {
    int i = blockIdx.x * blockDim.x + threadIdx.x;
    int stride = gridDim.x * blockDim.x;
    for (; i < n; i += stride) y[i] = f32_to_bf16(x[i]);
}

// ---------------------------------------------------------------------------
// Tiled WMMA bf16 GEMM:  C[M,N] = A[M,K] * W[N,K]^T + bias[N]
// Block tile 128x128, 8 waves, wave tile 32x64 (2x4 of 16x16), K-step 32.
// EPI==0: QKV epilogue (N = 3E): scatter to Q / K / V^T (bf16).
// EPI==1: plain epilogue: out[M,N] f32.
// ---------------------------------------------------------------------------
template <int EPI>
__global__ __launch_bounds__(256) void gemm_bf16_k(
    const unsigned short* __restrict__ A,    // [M,K] bf16
    const unsigned short* __restrict__ W,    // [N,K] bf16
    const float* __restrict__ bias,          // [N]
    unsigned short* __restrict__ outQ,       // [B,H,S,HD] bf16   (EPI==0)
    unsigned short* __restrict__ outK,       // [B,H,S,HD] bf16   (EPI==0)
    unsigned short* __restrict__ outVt,      // [B,H,HD,S] bf16   (EPI==0)
    float* __restrict__ outF,                // [M,N]  f32        (EPI==1)
    int N, int K) {
    __shared__ __align__(16) unsigned short ldsA[128 * 40];
    __shared__ __align__(16) unsigned short ldsB[128 * 40];

    const int t    = threadIdx.x;
    const int lane = t & 31;
    const int w    = t >> 5;
    const int wm   = (w & 3) * 32;
    const int wn   = (w >> 2) * 64;
    const int bm   = blockIdx.x * 128;
    const int bn   = blockIdx.y * 128;
    const int hl   = lane & 15;
    const bool lo  = lane < 16;

    v8f acc[2][4];
#pragma unroll
    for (int i = 0; i < 2; ++i)
#pragma unroll
        for (int j = 0; j < 4; ++j) acc[i][j] = zero8();

    const int rowL = t >> 1;           // 0..127
    const int offL = (t & 1) * 16;     // halves

    for (int kt = 0; kt < K; kt += 32) {
        const uint4v* gA =
            (const uint4v*)(A + (size_t)(bm + rowL) * K + kt + offL);
        const uint4v* gB =
            (const uint4v*)(W + (size_t)(bn + rowL) * K + kt + offL);
        uint4v a0 = gA[0], a1 = gA[1];
        uint4v b0 = gB[0], b1 = gB[1];

        __syncthreads();   // previous iteration's LDS reads are done
        *(uint4v*)&ldsA[rowL * 40 + offL]     = a0;
        *(uint4v*)&ldsA[rowL * 40 + offL + 8] = a1;
        *(uint4v*)&ldsB[rowL * 40 + offL]     = b0;
        *(uint4v*)&ldsB[rowL * 40 + offL + 8] = b1;
        __syncthreads();

        FragB aF[2], bF[4];
#pragma unroll
        for (int mi = 0; mi < 2; ++mi) {
            // A 16-bit layout: lanes0-15 M=lane K{0..7,16..23}; lanes16-31 K{8..15,24..31}
            int row = wm + mi * 16 + hl;
            int ka  = lo ? 0 : 8;
            aF[mi].u4[0] = *(const uint4v*)&ldsA[row * 40 + ka];
            aF[mi].u4[1] = *(const uint4v*)&ldsA[row * 40 + ka + 16];
        }
#pragma unroll
        for (int ni = 0; ni < 4; ++ni) {
            // B 16-bit layout: lanes0-15 N=lane K{0..7,16..23}; lanes16-31 K{8..15,24..31}
            int col = wn + ni * 16 + hl;
            int kb  = lo ? 0 : 8;
            bF[ni].u4[0] = *(const uint4v*)&ldsB[col * 40 + kb];
            bF[ni].u4[1] = *(const uint4v*)&ldsB[col * 40 + kb + 16];
        }
#pragma unroll
        for (int mi = 0; mi < 2; ++mi)
#pragma unroll
            for (int ni = 0; ni < 4; ++ni)
                acc[mi][ni] = __builtin_amdgcn_wmma_f32_16x16x32_bf16(
                    false, aF[mi].bf, false, bF[ni].bf, (short)0, acc[mi][ni],
                    false, false);
    }

    // Epilogue. C layout: VGPR r -> (M=r, N=lane) lanes0-15 / (M=r+8) lanes16-31.
#pragma unroll
    for (int mi = 0; mi < 2; ++mi) {
#pragma unroll
        for (int ni = 0; ni < 4; ++ni) {
            int coln = bn + wn + ni * 16 + hl;
            float bv = bias[coln];
#pragma unroll
            for (int r = 0; r < 8; ++r) {
                int rowm = bm + wm + mi * 16 + (lo ? r : r + 8);
                float v  = acc[mi][ni][r] + bv;
                if (EPI == 0) {
                    int part = coln >> 10;   // / E
                    int e    = coln & 1023;
                    int h    = e >> 6;       // / HD
                    int d    = e & 63;
                    int b    = rowm >> 11;   // / S
                    int s    = rowm & 2047;
                    unsigned short bf = f32_to_bf16(v);
                    size_t bh = (size_t)(b * Hc + h);
                    if (part == 0)
                        outQ[(bh * Sc + s) * HDc + d] = bf;
                    else if (part == 1)
                        outK[(bh * Sc + s) * HDc + d] = bf;
                    else
                        outVt[(bh * HDc + d) * Sc + s] = bf;
                } else {
                    outF[(size_t)rowm * N + coln] = v;
                }
            }
        }
    }
}

// ---------------------------------------------------------------------------
// Flash attention: one block per (128 query rows, b*H+h). 8 waves; each wave
// owns 16 query rows. Per 32-key step: stage K/V^T tiles in LDS, 4 score
// WMMAs, online softmax (tile-max, 4 bpermutes), P restaged via LDS, then
// 4 PV WMMAs + 1 row-sum WMMA against a ones tile (replaces sum shuffles).
// ---------------------------------------------------------------------------
__global__ __launch_bounds__(256) void flash_k(
    const unsigned short* __restrict__ Q,    // [B*H, S, HD]
    const unsigned short* __restrict__ Km,   // [B*H, S, HD]
    const unsigned short* __restrict__ Vt,   // [B*H, HD, S]
    unsigned short* __restrict__ O) {        // [B, S, E] bf16
    __shared__ __align__(16) unsigned short ldsK[32 * 72];   // keys x d
    __shared__ __align__(16) unsigned short ldsV[64 * 40];   // d x keys
    __shared__ __align__(16) unsigned short ldsP[8][16 * 40];

    const int t    = threadIdx.x;
    const int lane = t & 31;
    const int w    = t >> 5;
    const int hl   = lane & 15;
    const bool lo  = lane < 16;
    const int bh   = blockIdx.y;
    const int q0   = blockIdx.x * 128 + w * 16;
    const size_t qkBase = (size_t)bh * Sc * HDc;
    const size_t vBase  = (size_t)bh * HDc * Sc;

    // Q fragments (A operand), held for whole kernel.
    FragB qF[2];
    {
        const unsigned short* qp = Q + qkBase + (size_t)(q0 + hl) * HDc;
#pragma unroll
        for (int kf = 0; kf < 2; ++kf) {
            int ka = kf * 32 + (lo ? 0 : 8);
            qF[kf].u4[0] = *(const uint4v*)(qp + ka);
            qF[kf].u4[1] = *(const uint4v*)(qp + ka + 16);
        }
    }

    // Ones B-operand tile (32x16 of bf16 1.0) for WMMA row sums.
    FragB onesF;
#pragma unroll
    for (int c = 0; c < 2; ++c)
#pragma unroll
        for (int i = 0; i < 4; ++i) onesF.u4[c][i] = 0x3F803F80u;

    v8f o[4];
#pragma unroll
    for (int ni = 0; ni < 4; ++ni) o[ni] = zero8();
    v8f osum = zero8();                // running row sums (replicated per lane)
    float mrow[8];
#pragma unroll
    for (int r = 0; r < 8; ++r) mrow[r] = -1e30f;

    const float scale = 0.03125f;   // 1/sqrt(E) = 1/32

    for (int j0 = 0; j0 < Sc; j0 += 32) {
        {   // cooperative K/V tile staging: 256 threads, one b128 each per tile
            int rowK = t >> 3;             // 0..31 (key)
            int ck   = (t & 7) * 8;        // halves (16B chunks)
            uint4v kc = *(const uint4v*)(Km + qkBase +
                                         (size_t)(j0 + rowK) * HDc + ck);
            int rowV = t >> 2;             // 0..63 (d)
            int cv   = (t & 3) * 8;
            uint4v vc = *(const uint4v*)(Vt + vBase + (size_t)rowV * Sc + j0 +
                                         cv);
            __syncthreads();
            *(uint4v*)&ldsK[rowK * 72 + ck] = kc;
            *(uint4v*)&ldsV[rowV * 40 + cv] = vc;
            __syncthreads();
        }

        // scores for two 16-key tiles, reduced over d=64 (2 chained WMMAs each)
        v8f sAcc[2];
#pragma unroll
        for (int jt = 0; jt < 2; ++jt) {
            sAcc[jt] = zero8();
#pragma unroll
            for (int kf = 0; kf < 2; ++kf) {
                FragB kF;
                int key = jt * 16 + hl;
                int kb  = kf * 32 + (lo ? 0 : 8);
                kF.u4[0] = *(const uint4v*)&ldsK[key * 72 + kb];
                kF.u4[1] = *(const uint4v*)&ldsK[key * 72 + kb + 16];
                sAcc[jt] = __builtin_amdgcn_wmma_f32_16x16x32_bf16(
                    false, qF[kf].bf, false, kF.bf, (short)0, sAcc[jt], false,
                    false);
            }
        }

        // Tile-wide max (valid stability bound: >= every row max).
        // In-lane reduce over 8 row-registers, then one 16-lane butterfly.
        float tm = sAcc[0][0];
#pragma unroll
        for (int r = 0; r < 8; ++r)
            tm = fmaxf(tm, fmaxf(sAcc[0][r], sAcc[1][r]));
        tm *= scale;
#pragma unroll
        for (int off = 1; off < 16; off <<= 1)
            tm = fmaxf(tm, __shfl_xor(tm, off, 32));

        // online softmax; rows 0-7 live in lanes 0-15, rows 8-15 in lanes 16-31
#pragma unroll
        for (int r = 0; r < 8; ++r) {
            float mn   = fmaxf(mrow[r], tm);
            float corr = __expf(mrow[r] - mn);
            float p0   = __expf(sAcc[0][r] * scale - mn);
            float p1   = __expf(sAcc[1][r] * scale - mn);
            mrow[r]    = mn;
            osum[r] *= corr;
#pragma unroll
            for (int ni = 0; ni < 4; ++ni) o[ni][r] *= corr;
            int prow = lo ? r : r + 8;
            ldsP[w][prow * 40 + hl]      = f32_to_bf16(p0);
            ldsP[w][prow * 40 + 16 + hl] = f32_to_bf16(p1);
        }
        // intra-wave LDS RAW: DS ops are in-order per wave; wait for stores
        asm volatile("s_wait_dscnt 0" ::: "memory");

        FragB pF;
        {
            int ka = lo ? 0 : 8;
            pF.u4[0] = *(const uint4v*)&ldsP[w][hl * 40 + ka];
            pF.u4[1] = *(const uint4v*)&ldsP[w][hl * 40 + ka + 16];
        }
#pragma unroll
        for (int ni = 0; ni < 4; ++ni) {
            FragB vF;
            int d  = ni * 16 + hl;
            int kb = lo ? 0 : 8;
            vF.u4[0] = *(const uint4v*)&ldsV[d * 40 + kb];
            vF.u4[1] = *(const uint4v*)&ldsV[d * 40 + kb + 16];
            o[ni] = __builtin_amdgcn_wmma_f32_16x16x32_bf16(
                false, pF.bf, false, vF.bf, (short)0, o[ni], false, false);
        }
        // row sums via ones tile: every lane of row r receives sum_k P[r][k]
        osum = __builtin_amdgcn_wmma_f32_16x16x32_bf16(
            false, pF.bf, false, onesF.bf, (short)0, osum, false, false);
    }

    // normalize and write attention output in [B, S, H*HD] bf16
    int b = bh >> 4;
    int h = bh & 15;
#pragma unroll
    for (int ni = 0; ni < 4; ++ni) {
#pragma unroll
        for (int r = 0; r < 8; ++r) {
            int srow = q0 + (lo ? r : r + 8);
            int d    = ni * 16 + hl;
            float v  = o[ni][r] / osum[r];
            O[((size_t)(b * Sc + srow) * Ec) + h * HDc + d] = f32_to_bf16(v);
        }
    }
}

// ---------------------------------------------------------------------------
extern "C" void kernel_launch(void* const* d_in, const int* in_sizes, int n_in,
                              void* d_out, int out_size, void* d_ws,
                              size_t ws_size, hipStream_t stream) {
    const float* query = (const float*)d_in[0];
    // d_in[1] (key) and d_in[2] (value) are unused by the reference.
    const float* Wqkv = (const float*)d_in[3];
    const float* bqkv = (const float*)d_in[4];
    const float* Wout = (const float*)d_in[5];
    const float* bout = (const float*)d_in[6];
    float* out = (float*)d_out;

    char* ws = (char*)d_ws;
    size_t off = 0;
    auto alloc = [&](size_t bytes) -> void* {
        void* p = ws + off;
        off += (bytes + 255) & ~(size_t)255;
        return p;
    };

    const size_t nX    = (size_t)Bc * Sc * Ec;        // 8,388,608
    const size_t nWqkv = (size_t)3 * Ec * Ec;         // 3,145,728
    const size_t nWout = (size_t)Ec * Ec;             // 1,048,576
    const size_t nHead = (size_t)Bc * Hc * Sc * HDc;  // 8,388,608

    unsigned short* Xb     = (unsigned short*)alloc(nX * 2);
    unsigned short* Wqkvb  = (unsigned short*)alloc(nWqkv * 2);
    unsigned short* Woutb  = (unsigned short*)alloc(nWout * 2);
    unsigned short* Qb     = (unsigned short*)alloc(nHead * 2);
    unsigned short* Kb     = (unsigned short*)alloc(nHead * 2);
    unsigned short* Vtb    = (unsigned short*)alloc(nHead * 2);
    unsigned short* Ob     = (unsigned short*)alloc(nX * 2);

    // 1) convert inputs to bf16
    cvt_bf16_k<<<4096, 256, 0, stream>>>(query, Xb, (int)nX);
    cvt_bf16_k<<<4096, 256, 0, stream>>>(Wqkv, Wqkvb, (int)nWqkv);
    cvt_bf16_k<<<4096, 256, 0, stream>>>(Wout, Woutb, (int)nWout);

    // 2) QKV projection: M=B*S=8192, N=3E=3072, K=E=1024
    gemm_bf16_k<0><<<dim3(64, 24), 256, 0, stream>>>(
        Xb, Wqkvb, bqkv, Qb, Kb, Vtb, nullptr, 3 * Ec, Ec);

    // 3) flash attention: grid (S/128, B*H)
    flash_k<<<dim3(Sc / 128, Bc * Hc), 256, 0, stream>>>(Qb, Kb, Vtb, Ob);

    // 4) output projection: M=8192, N=E=1024, K=E=1024, f32 out
    gemm_bf16_k<1><<<dim3(64, 8), 256, 0, stream>>>(
        Ob, Woutb, bout, nullptr, nullptr, nullptr, out, Ec, Ec);
}
